// GNN_11269994185294
// MI455X (gfx1250) — compile-verified, matrix operand order checked by reading, not run on hardware
//
#include <hip/hip_runtime.h>
#include <math.h>

typedef __attribute__((ext_vector_type(16))) __bf16 v16bf;
typedef __attribute__((ext_vector_type(8)))  __bf16 v8bf;
typedef __attribute__((ext_vector_type(8)))  float  v8f;
typedef int v4i __attribute__((vector_size(16)));                 // matches builtin's V4i
typedef __attribute__((address_space(1))) v4i gv4i;               // global
typedef __attribute__((address_space(3))) v4i lv4i;               // LDS

#define TILE_LD 40   // padded LDS row stride (elements) -> conflict-free ds_load_b128

#if defined(__gfx1250__) && __has_builtin(__builtin_amdgcn_global_load_async_to_lds_b128)
#define USE_ASYNC 1
#else
#define USE_ASYNC 0
#endif

// ---------------- helper kernels ----------------

__global__ void cvt_f32_bf16_kernel(const float* __restrict__ s, __bf16* __restrict__ d, size_t n) {
    size_t i  = (size_t)blockIdx.x * blockDim.x + threadIdx.x;
    size_t st = (size_t)gridDim.x * blockDim.x;
    for (; i < n; i += st) d[i] = (__bf16)s[i];
}

// d[c*R + r] = (bf16) s[r*C + c]   : convert + transpose  (s is [R][C], d is [C][R])
__global__ void cvt_transpose_kernel(const float* __restrict__ s, __bf16* __restrict__ d,
                                     int R, int C) {
    size_t n  = (size_t)R * C;
    size_t i  = (size_t)blockIdx.x * blockDim.x + threadIdx.x;
    size_t st = (size_t)gridDim.x * blockDim.x;
    for (; i < n; i += st) {
        size_t r = i / C; int c = (int)(i - r * C);
        d[(size_t)c * R + r] = (__bf16)s[i];
    }
}

// nfT[c][r] = c < dIn ? nx[r][c] : ny[r][c-dIn]   (transposed concat -> [dz][Nn] bf16)
__global__ void pack_nf_t_kernel(const float* __restrict__ nx, const float* __restrict__ ny,
                                 __bf16* __restrict__ d, int Nn, int dIn, int dz) {
    size_t n  = (size_t)Nn * dz;
    size_t i  = (size_t)blockIdx.x * blockDim.x + threadIdx.x;
    size_t st = (size_t)gridDim.x * blockDim.x;
    int big = dz - dIn;
    for (; i < n; i += st) {
        size_t r = i / dz; int c = (int)(i - r * dz);
        float v = (c < dIn) ? nx[r * dIn + c] : ny[r * (size_t)big + (c - dIn)];
        d[(size_t)c * Nn + r] = (__bf16)v;
    }
}

__global__ void init_degree_kernel(float* __restrict__ deg, int n) {
    int i = blockIdx.x * blockDim.x + threadIdx.x;
    if (i < n) deg[i] = 1.0f;
}

// ---------------- tile staging ----------------
// Stage a 128x32 bf16 tile (row-major, ld elements) into LDS (stride TILE_LD).
// All shapes divide exactly -> no bounds checks, straight b128 traffic.
__device__ __forceinline__ void stage_tile(__bf16* sm, const __bf16* g,
                                           int rowBase, int ld, int k0, int t) {
#pragma unroll
    for (int it = 0; it < 2; ++it) {
        int id = t + it * 256;      // 0..511
        int r  = id >> 2;           // 0..127
        int c8 = (id & 3) << 3;     // 0,8,16,24
        const __bf16* src = g + (size_t)(rowBase + r) * ld + k0 + c8;
        __bf16*       dst = sm + r * TILE_LD + c8;
#if USE_ASYNC
        __builtin_amdgcn_global_load_async_to_lds_b128((gv4i*)src, (lv4i*)dst, 0, 0);
#else
        *(uint4*)dst = *(const uint4*)src;
#endif
    }
}

__device__ __forceinline__ void wait_async_done() {
#if USE_ASYNC
#if __has_builtin(__builtin_amdgcn_s_wait_asynccnt)
    __builtin_amdgcn_s_wait_asynccnt(0);
#else
    asm volatile("s_wait_asynccnt 0x0" ::: "memory");
#endif
#endif
}

// A/B fragment load, 16x32 bf16 tile starting at LDS row `row0`.
// ISA layout: lane<16 -> row=lane,   K 0..7 in h[0], K 16..23 in h[1]
//             lane>=16-> row=lane-16,K 8..15 in h[0], K 24..31 in h[1]
__device__ __forceinline__ v16bf load_frag(const __bf16* sm, int row0, int lane) {
    int r  = row0 + (lane & 15);
    int kh = (lane < 16) ? 0 : 8;
    union { v16bf v; v8bf h[2]; } u;
    u.h[0] = *(const v8bf*)(sm + r * TILE_LD + kh);
    u.h[1] = *(const v8bf*)(sm + r * TILE_LD + kh + 16);
    return u.v;
}

// ---------------- main GEMM ----------------
// C[M,N] = epilogue( A[M,K](bf16,row-major) @ B^T ), B stored [N,K] bf16 row-major.
// Block tile 128x128, 8 waves as 2(M) x 4(N), each wave 64x32 = 4x2 WMMA tiles.
// Double-buffered LDS, async global->LDS staging overlapped with WMMA.
// EPI: 0 = +bias[col] -> bf16
//      1 = exp(acc*invTemp) -> bf16, row-sums atomicAdd'ed into degree[]
//      2 = (target_feature + acc)/degree[row] -> bf16
//      3 = +bias[col] -> float (final output)
template <int EPI>
__global__ __launch_bounds__(256) void gemm_bf16_wmma(
    const __bf16* __restrict__ A, const __bf16* __restrict__ B, void* __restrict__ Cv,
    int M, int N, int K, int lda, int ldb, int ldc,
    const float* __restrict__ bias, float* __restrict__ degree,
    const float* __restrict__ tx, int dimIn, float invTemp) {
    __shared__ __bf16 sA[2][128 * TILE_LD];
    __shared__ __bf16 sB[2][128 * TILE_LD];

    const int t     = threadIdx.x;
    const int lane  = t & 31;
    const int wave  = t >> 5;
    const int waveM = (wave >> 2) * 64;  // 2 waves along M
    const int waveN = (wave & 3) * 32;   // 4 waves along N
    const int mBase = blockIdx.y * 128;
    const int nBase = blockIdx.x * 128;

    v8f acc[4][2];
#pragma unroll
    for (int i = 0; i < 4; ++i)
#pragma unroll
        for (int j = 0; j < 2; ++j)
            acc[i][j] = (v8f){0.f, 0.f, 0.f, 0.f, 0.f, 0.f, 0.f, 0.f};

    // prologue: stage first K-tile into buffer 0
    stage_tile(sA[0], A, mBase, lda, 0, t);
    stage_tile(sB[0], B, nBase, ldb, 0, t);

    int p = 0;
    for (int k0 = 0; k0 < K; k0 += 32) {
        wait_async_done();
        __syncthreads();                 // tile k0 visible to all waves; prev buf free
        if (k0 + 32 < K) {               // stage next tile into the other buffer
            stage_tile(sA[p ^ 1], A, mBase, lda, k0 + 32, t);
            stage_tile(sB[p ^ 1], B, nBase, ldb, k0 + 32, t);
        }

        v16bf aF[4], bF[2];
#pragma unroll
        for (int i = 0; i < 4; ++i) aF[i] = load_frag(sA[p], waveM + i * 16, lane);
#pragma unroll
        for (int j = 0; j < 2; ++j) bF[j] = load_frag(sB[p], waveN + j * 16, lane);
#pragma unroll
        for (int i = 0; i < 4; ++i)
#pragma unroll
            for (int j = 0; j < 2; ++j)
                acc[i][j] = __builtin_amdgcn_wmma_f32_16x16x32_bf16(
                    false, aF[i], false, bF[j], (short)0, acc[i][j], false, false);
        p ^= 1;
    }

    // Epilogue.  C layout: vgpr g, lane<16 -> row g, lane>=16 -> row g+8; col = lane&15.
    const int colIn   = lane & 15;
    const int rowHalf = (lane < 16) ? 0 : 8;
#pragma unroll
    for (int i = 0; i < 4; ++i) {
#pragma unroll
        for (int g = 0; g < 8; ++g) {
            int row = mBase + waveM + i * 16 + rowHalf + g;
            if (EPI == 1) {
                float s = 0.f;
#pragma unroll
                for (int j = 0; j < 2; ++j) {
                    int col  = nBase + waveN + j * 16 + colIn;
                    float e  = __expf(acc[i][j][g] * invTemp);
                    ((__bf16*)Cv)[(size_t)row * ldc + col] = (__bf16)e;
                    s += e;
                }
#pragma unroll
                for (int m = 1; m < 16; m <<= 1) s += __shfl_xor(s, m, 32);
                if (colIn == 0) atomicAdd(&degree[row], s);
            } else {
#pragma unroll
                for (int j = 0; j < 2; ++j) {
                    int col = nBase + waveN + j * 16 + colIn;
                    float v = acc[i][j][g];
                    if (EPI == 0) {
                        v += bias[col];
                        ((__bf16*)Cv)[(size_t)row * ldc + col] = (__bf16)v;
                    } else if (EPI == 2) {
                        float tf = (col < dimIn) ? tx[(size_t)row * dimIn + col] : 0.f;
                        v = (tf + v) / degree[row];
                        ((__bf16*)Cv)[(size_t)row * ldc + col] = (__bf16)v;
                    } else {
                        v += bias[col];
                        ((float*)Cv)[(size_t)row * ldc + col] = v;
                    }
                }
            }
        }
    }
}

// ---------------- orchestration ----------------

extern "C" void kernel_launch(void* const* d_in, const int* in_sizes, int n_in,
                              void* d_out, int out_size, void* d_ws, size_t ws_size,
                              hipStream_t stream) {
    const float* nx = (const float*)d_in[0];   // neighbor_x [Nn, DIN]
    const float* ny = (const float*)d_in[1];   // neighbor_y [Nn, BIG]
    const float* tx = (const float*)d_in[2];   // target_x   [Nt, DIN]
    const float* Wq = (const float*)d_in[3];   // [DIN, DZ]
    const float* bq = (const float*)d_in[4];   // [DZ]
    const float* Wk = (const float*)d_in[5];   // [DIN, DZ]
    const float* bk = (const float*)d_in[6];   // [DZ]
    const float* wW = (const float*)d_in[7];   // [DZ, DZ]
    const float* wb = (const float*)d_in[8];   // [DZ]
    const float* oW = (const float*)d_in[9];   // [DZ, BIG]
    const float* ob = (const float*)d_in[10];  // [BIG]

    const int DZ  = in_sizes[4];
    const int BIG = in_sizes[10];
    const int DIN = in_sizes[3] / DZ;
    const int Nn  = in_sizes[0] / DIN;
    const int Nt  = in_sizes[2] / DIN;
    const float invTemp = 1.0f / sqrtf((float)DZ);

    // workspace carve-out: A operands row-major, all B operands pre-transposed [N][K]
    size_t off = 0;
    auto alloc = [&](size_t bytes) -> char* {
        off = (off + 255) & ~(size_t)255;
        char* p = (char*)d_ws + off;
        off += bytes;
        return p;
    };
    __bf16* txB  = (__bf16*)alloc((size_t)Nt * DIN * 2);   // [Nt, DIN]
    __bf16* nxB  = (__bf16*)alloc((size_t)Nn * DIN * 2);   // [Nn, DIN]
    __bf16* WqT  = (__bf16*)alloc((size_t)DIN * DZ * 2);   // [DZ, DIN]
    __bf16* WkT  = (__bf16*)alloc((size_t)DIN * DZ * 2);   // [DZ, DIN]
    __bf16* wWT  = (__bf16*)alloc((size_t)DZ * DZ * 2);    // [DZ, DZ] (transposed)
    __bf16* oWT  = (__bf16*)alloc((size_t)DZ * BIG * 2);   // [BIG, DZ]
    __bf16* nfT  = (__bf16*)alloc((size_t)Nn * DZ * 2);    // [DZ, Nn]
    __bf16* qB   = (__bf16*)alloc((size_t)Nt * DZ * 2);    // [Nt, DZ]
    __bf16* kB   = (__bf16*)alloc((size_t)Nn * DZ * 2);    // [Nn, DZ]  (acts as K^T tile source)
    __bf16* simB = (__bf16*)alloc((size_t)Nt * Nn * 2);    // [Nt, Nn]
    __bf16* aggB = (__bf16*)alloc((size_t)Nt * DZ * 2);    // [Nt, DZ]
    __bf16* hB   = (__bf16*)alloc((size_t)Nt * DZ * 2);    // [Nt, DZ]
    float*  deg  = (float*)alloc((size_t)Nt * 4);
    (void)ws_size;

    auto nblk = [](size_t n) { return (int)(((n + 255) / 256 < 4096) ? (n + 255) / 256 : 4096); };
    cvt_f32_bf16_kernel<<<nblk((size_t)Nt * DIN), 256, 0, stream>>>(tx, txB, (size_t)Nt * DIN);
    cvt_f32_bf16_kernel<<<nblk((size_t)Nn * DIN), 256, 0, stream>>>(nx, nxB, (size_t)Nn * DIN);
    cvt_transpose_kernel<<<nblk((size_t)DIN * DZ), 256, 0, stream>>>(Wq, WqT, DIN, DZ);
    cvt_transpose_kernel<<<nblk((size_t)DIN * DZ), 256, 0, stream>>>(Wk, WkT, DIN, DZ);
    cvt_transpose_kernel<<<nblk((size_t)DZ * DZ), 256, 0, stream>>>(wW, wWT, DZ, DZ);
    cvt_transpose_kernel<<<nblk((size_t)DZ * BIG), 256, 0, stream>>>(oW, oWT, DZ, BIG);
    pack_nf_t_kernel<<<4096, 256, 0, stream>>>(nx, ny, nfT, Nn, DIN, DZ);
    init_degree_kernel<<<(Nt + 255) / 256, 256, 0, stream>>>(deg, Nt);

    dim3 blk(256);
    auto grid = [](int M, int N) { return dim3((unsigned)(N / 128), (unsigned)(M / 128)); };

    // q = tx @ Wq + bq          [Nt, DZ]
    gemm_bf16_wmma<0><<<grid(Nt, DZ), blk, 0, stream>>>(
        txB, WqT, qB, Nt, DZ, DIN, DIN, DIN, DZ, bq, nullptr, nullptr, 0, 0.f);
    // k = nx @ Wk + bk          [Nn, DZ]
    gemm_bf16_wmma<0><<<grid(Nn, DZ), blk, 0, stream>>>(
        nxB, WkT, kB, Nn, DZ, DIN, DIN, DIN, DZ, bk, nullptr, nullptr, 0, 0.f);
    // sim = exp(q @ k^T / temp) [Nt, Nn]; degree += rowsum(sim)
    gemm_bf16_wmma<1><<<grid(Nt, Nn), blk, 0, stream>>>(
        qB, kB, simB, Nt, Nn, DZ, DZ, DZ, Nn, nullptr, deg, nullptr, 0, invTemp);
    // agg = (tf + sim @ NF) / degree   [Nt, DZ]
    gemm_bf16_wmma<2><<<grid(Nt, DZ), blk, 0, stream>>>(
        simB, nfT, aggB, Nt, DZ, Nn, Nn, Nn, DZ, nullptr, deg, tx, DIN, 0.f);
    // h = agg @ wW + wb         [Nt, DZ]
    gemm_bf16_wmma<0><<<grid(Nt, DZ), blk, 0, stream>>>(
        aggB, wWT, hB, Nt, DZ, DZ, DZ, DZ, DZ, wb, nullptr, nullptr, 0, 0.f);
    // out = h @ oW + ob         [Nt, BIG]  (fp32 output)
    gemm_bf16_wmma<3><<<grid(Nt, BIG), blk, 0, stream>>>(
        hB, oWT, d_out, Nt, BIG, DZ, DZ, DZ, BIG, ob, nullptr, nullptr, 0, 0.f);
}
